// SymmetricCGAAttention_30288109371520
// MI455X (gfx1250) — matrix operand order repaired and checked by reading, not compiled, not run on hardware
//
#include <hip/hip_runtime.h>

// ---------------------------------------------------------------------------
// Problem constants (match reference)
// ---------------------------------------------------------------------------
#define BATCH   8
#define SEQ     1025
#define DMODEL  1024
#define HEADS   16
#define DHEAD   64
#define NPAD    1056            // 33 * 32, covers all 32-wide key chunks
#define NQT     65              // ceil(1025/16) query tiles
#define MPAD    8256            // 129 * 64 token rows (8*1025 = 8200 valid)
#define MBLK    129             // 64-row GEMM blocks
#define EPSF    1e-8f

// GEMM tiling
#define BN      128             // N columns per block
#define LDSROW  40              // LDS row stride in bf16 elems (32 data + 8 pad)

typedef __bf16 bf16_t;
typedef __attribute__((ext_vector_type(16))) __bf16 v16bf;
typedef __attribute__((ext_vector_type(8)))  __bf16 v8bf;
typedef __attribute__((ext_vector_type(8)))  float  v8f;
typedef __attribute__((ext_vector_type(4)))  unsigned int u32x4;
typedef __attribute__((ext_vector_type(8)))  int i32x8;
typedef __attribute__((ext_vector_type(4)))  int i32x4;

// ---------------------------------------------------------------------------
// WMMA helpers (contiguous k = 16*(lane/16) + s in A and B, wave32)
// ---------------------------------------------------------------------------
__device__ __forceinline__ v8f wmma_bf16(v16bf a, v16bf b, v8f c) {
  return __builtin_amdgcn_wmma_f32_16x16x32_bf16(false, a, false, b, (short)0, c,
                                                 false, false);
}

__device__ __forceinline__ v16bf load_frag_2x8(const bf16_t* p0, const bf16_t* p1) {
  v8bf lo = *(const v8bf*)p0;           // 16B load
  v8bf hi = *(const v8bf*)p1;           // 16B load
  v16bf r;
#pragma unroll
  for (int i = 0; i < 8; ++i) { r[i] = lo[i]; r[8 + i] = hi[i]; }
  return r;
}

// A-matrix 16x32 (MxK), row-major source, leading dim ld.
__device__ __forceinline__ v16bf load_a_frag(const bf16_t* A, int ld, int lane) {
  const int m = lane & 15, kh = lane >> 4;
  const bf16_t* p = A + (size_t)m * ld + kh * 16;
  return load_frag_2x8(p, p + 8);
}

// B-matrix 32x16 (KxN) from row-major W[N][K] (B = W^T chunk), global memory.
__device__ __forceinline__ v16bf load_bT_frag(const bf16_t* Wrow0, int ld, int lane) {
  const int n = lane & 15, kh = lane >> 4;
  const bf16_t* p = Wrow0 + (size_t)n * ld + kh * 16;
  return load_frag_2x8(p, p + 8);
}

// B-matrix 32x16 from row-major V[key][d] (K=key, N=d). Strided.
__device__ __forceinline__ v16bf load_bV_frag(const bf16_t* Vchunk, int d0, int ld,
                                              int lane) {
  const int d = d0 + (lane & 15), kh = lane >> 4;
  const bf16_t* p = Vchunk + (size_t)(kh * 16) * ld + d;
  v16bf r;
#pragma unroll
  for (int i = 0; i < 16; ++i) r[i] = p[(size_t)i * ld];
  return r;
}

// ---------------------------------------------------------------------------
// Tensor Data Mover: async 2D tile load (global -> LDS), bf16 elements.
// Descriptor per CDNA5 ISA 08_async_tensor §8.3-8.6.
// ---------------------------------------------------------------------------
__device__ __forceinline__ void tdm_load_2d_bf16(unsigned lds_off, const void* gptr,
                                                 unsigned tile_d0, unsigned tile_d1,
                                                 unsigned long long stride0) {
  const unsigned long long ga = (unsigned long long)(size_t)gptr;
  u32x4 g0;
  g0[0] = 1u;                                       // count=1, user descriptor
  g0[1] = lds_off;                                  // lds_addr (bytes)
  g0[2] = (unsigned)(ga & 0xFFFFFFFFu);             // global_addr[31:0]
  g0[3] = (unsigned)((ga >> 32) & 0x01FFFFFFu)      // global_addr[56:32]
          | (2u << 30);                             // type = 2 ("image")
  i32x8 g1;
  // word0: workgroup_mask=0 | data_size=1 (2B) | pad_enable, interval=16 DW (code 3),
  //        amount=4 DW (code 3)  -> LDS row stride = 64B + 16B = 80B
  g1[0] = (int)((1u << 16) | (1u << 20) | (3u << 22) | (3u << 25));
  g1[1] = (int)((tile_d0 & 0xFFFFu) << 16);         // abar=0 | tensor_dim0[15:0]
  g1[2] = (int)(((tile_d0 >> 16) & 0xFFFFu) | ((tile_d1 & 0xFFFFu) << 16));
  g1[3] = (int)(((tile_d1 >> 16) & 0xFFFFu) | ((tile_d0 & 0xFFFFu) << 16)); // tile_dim0
  g1[4] = (int)(tile_d1 & 0xFFFFu);                 // tile_dim1 | tile_dim2=0
  g1[5] = (int)(unsigned)(stride0 & 0xFFFFFFFFu);   // tensor_dim0_stride[31:0]
  g1[6] = (int)(unsigned)((stride0 >> 32) & 0xFFFFu); // stride0[47:32] | stride1_lo=0
  g1[7] = 0;                                        // stride1 hi
  i32x4 g2 = {};                                    // dims 2/3 unused (zeros)
  i32x4 g3 = {};
#if defined(__clang_major__) && __clang_major__ >= 23
  i32x8 g4 = {};
  __builtin_amdgcn_tensor_load_to_lds(g0, g1, g2, g3, g4, 0);
#else
  __builtin_amdgcn_tensor_load_to_lds(g0, g1, g2, g3, 0);
#endif
}

// ---------------------------------------------------------------------------
// Kernel 0: f32 -> bf16 conversion (weights)
// ---------------------------------------------------------------------------
__global__ void k_cvt_bf16(const float* __restrict__ src, bf16_t* __restrict__ dst,
                           int n) {
  int i = blockIdx.x * blockDim.x + threadIdx.x;
  if (i < n) dst[i] = (bf16_t)src[i];
}

// ---------------------------------------------------------------------------
// Kernel 1: zero-centered RMSNorm -> h (bf16); zero pad rows of h and y
// ---------------------------------------------------------------------------
__device__ __forceinline__ float block_sum256(float v, float* sm) {
#pragma unroll
  for (int o = 16; o > 0; o >>= 1) v += __shfl_xor(v, o);
  const int wid = threadIdx.x >> 5, lid = threadIdx.x & 31;
  if (lid == 0) sm[wid] = v;
  __syncthreads();
  float r = 0.f;
  if (threadIdx.x < 8) {
    r = sm[threadIdx.x];
#pragma unroll
    for (int o = 4; o > 0; o >>= 1) r += __shfl_xor(r, o);
    if (threadIdx.x == 0) sm[0] = r;
  }
  __syncthreads();
  r = sm[0];
  __syncthreads();
  return r;
}

__global__ void k_zc_rmsnorm(const float* __restrict__ x, const float* __restrict__ g,
                             bf16_t* __restrict__ h, bf16_t* __restrict__ ypad) {
  __shared__ float sm[8];
  const int row = blockIdx.x;
  const int t = threadIdx.x;
  if (row >= BATCH * SEQ) {  // pad rows: zero h and y so GEMMs see clean zeros
    for (int c = t; c < DMODEL; c += 256) {
      h[(size_t)row * DMODEL + c] = (bf16_t)0.f;
      ypad[(size_t)row * DMODEL + c] = (bf16_t)0.f;
    }
    return;
  }
  const float* xr = x + (size_t)row * DMODEL;
  float v[4];
  float s = 0.f;
#pragma unroll
  for (int i = 0; i < 4; ++i) { v[i] = xr[t + 256 * i]; s += v[i]; }
  const float mean = block_sum256(s, sm) * (1.0f / DMODEL);
  float s2 = 0.f;
#pragma unroll
  for (int i = 0; i < 4; ++i) { float d = v[i] - mean; s2 += d * d; }
  const float var = block_sum256(s2, sm) * (1.0f / DMODEL);
  const float rs = rsqrtf(var + EPSF);
#pragma unroll
  for (int i = 0; i < 4; ++i) {
    const int c = t + 256 * i;
    h[(size_t)row * DMODEL + c] = (bf16_t)((v[i] - mean) * rs * g[c]);
  }
}

// ---------------------------------------------------------------------------
// Kernel 2: bf16 WMMA GEMM with TDM-staged weights.
//   C[M,Nc] = A[M,K] * W[Nc,K]^T + bias (f32 out)
// Block = 4 waves, computes 64(M) x 128(N). Weight tile (128 x 32 bf16) is
// DMA'd into LDS by the Tensor Data Mover, double-buffered, shared by all
// waves. All 8 B fragments are loaded from LDS up front (one big ds clause)
// so the 8 WMMAs retire with partial DS waits instead of a full stall each.
// ---------------------------------------------------------------------------
__global__ __launch_bounds__(128) void k_gemm_bf16(
    const bf16_t* __restrict__ A, const bf16_t* __restrict__ W,
    const float* __restrict__ bias, float* __restrict__ C, int K, int Nc) {
  __shared__ __align__(16) bf16_t Bt[2][BN * LDSROW];
  const int lane = threadIdx.x & 31;
  const int wave = threadIdx.x >> 5;
  const int m0 = blockIdx.y * 64 + wave * 16;
  const int n0 = blockIdx.x * BN;
  v8f acc[8] = {};
  const bf16_t* Arow = A + (size_t)m0 * K;
  const bf16_t* Wt = W + (size_t)n0 * K;

  if (wave == 0) {
    tdm_load_2d_bf16((unsigned)(size_t)&Bt[0][0], Wt, 32u, (unsigned)BN,
                     (unsigned long long)K);
    __builtin_amdgcn_s_wait_tensorcnt(0);
  }
  __syncthreads();

  const int kh16 = (lane >> 4) * 16;
  const int nl = lane & 15;
  for (int k0 = 0; k0 < K; k0 += 32) {
    const int cur = (k0 >> 5) & 1;
    const bool more = (k0 + 32) < K;
    if (wave == 0 && more) {
      tdm_load_2d_bf16((unsigned)(size_t)&Bt[cur ^ 1][0], Wt + k0 + 32, 32u,
                       (unsigned)BN, (unsigned long long)K);
    }
    __builtin_prefetch(Arow + (size_t)nl * K + k0 + 64, 0, 1);
    const v16bf a = load_a_frag(Arow + k0, K, lane);
    const bf16_t* Bb = &Bt[cur][0];
    // issue all 16 ds_load_b128 first, then the 8 WMMAs
    v16bf bfrag[8];
#pragma unroll
    for (int j = 0; j < 8; ++j) {
      const bf16_t* p = Bb + (size_t)(16 * j + nl) * LDSROW + kh16;
      bfrag[j] = load_frag_2x8(p, p + 8);
    }
#pragma unroll
    for (int j = 0; j < 8; ++j) acc[j] = wmma_bf16(a, bfrag[j], acc[j]);
    if (wave == 0 && more) __builtin_amdgcn_s_wait_tensorcnt(0);
    __syncthreads();
  }

  const int mh = (lane >> 4) * 8;
#pragma unroll
  for (int j = 0; j < 8; ++j) {
    const float bv = bias[n0 + 16 * j + nl];
#pragma unroll
    for (int r = 0; r < 8; ++r)
      C[(size_t)(m0 + mh + r) * Nc + n0 + 16 * j + nl] = acc[j][r] + bv;
  }
}

// ---------------------------------------------------------------------------
// Kernel 3: RoPE + split to Q/K/V in [B,H,NPAD,DHEAD] bf16; zero pad keys
// ---------------------------------------------------------------------------
__global__ void k_rope_split(const float* __restrict__ qkv, const float* __restrict__ fc,
                             bf16_t* __restrict__ Q, bf16_t* __restrict__ Km,
                             bf16_t* __restrict__ V) {
  const size_t idx = (size_t)blockIdx.x * blockDim.x + threadIdx.x;
  const size_t total = (size_t)BATCH * NPAD * (3 * DMODEL);
  if (idx >= total) return;
  const int c = (int)(idx % (3 * DMODEL));
  const size_t tn = idx / (3 * DMODEL);
  const int n = (int)(tn % NPAD);
  const int b = (int)(tn / NPAD);
  const int part = c >> 10;      // 0=q 1=k 2=v
  const int cc = c & 1023;
  const int hh = cc >> 6;
  const int d = cc & 63;
  bf16_t* dst = (part == 0) ? Q : (part == 1) ? Km : V;
  const size_t o = ((size_t)(b * HEADS + hh) * NPAD + n) * DHEAD + d;
  if (n >= SEQ) { dst[o] = (bf16_t)0.f; return; }
  const size_t row = (size_t)(b * SEQ + n) * (3 * DMODEL);
  float val;
  if (part == 2) {
    val = qkv[row + c];
  } else {
    const int d2 = d >> 1;
    const float fr = fc[(size_t)n * DHEAD + d2 * 2 + 0];
    const float fi = fc[(size_t)n * DHEAD + d2 * 2 + 1];
    const float tr = qkv[row + part * DMODEL + hh * DHEAD + (d & ~1)];
    const float ti = qkv[row + part * DMODEL + hh * DHEAD + (d | 1)];
    val = (d & 1) ? (tr * fi + ti * fr) : (tr * fr - ti * fi);
  }
  dst[o] = (bf16_t)val;
}

// ---------------------------------------------------------------------------
// Kernel 4: flash attention, one wave per (b, h, 16-query tile).
// K^T fragments hoisted ahead of the score WMMAs; V fragments issued right
// after the score WMMAs so their global latency hides under softmax VALU
// work and the LDS transpose.
// ---------------------------------------------------------------------------
__global__ __launch_bounds__(32) void k_attention(
    const bf16_t* __restrict__ Q, const bf16_t* __restrict__ Km,
    const bf16_t* __restrict__ V, const float* __restrict__ Bbias,
    bf16_t* __restrict__ Y) {
  __shared__ __align__(16) bf16_t Pst[16 * 32];
  const int lane = threadIdx.x;
  const int qt = blockIdx.x, hh = blockIdx.y, b = blockIdx.z;
  const size_t headoff = (size_t)(b * HEADS + hh) * NPAD * DHEAD;
  const bf16_t* Qb = Q + headoff + (size_t)qt * 16 * DHEAD;
  const bf16_t* Kb = Km + headoff;
  const bf16_t* Vb = V + headoff;

  const v16bf aq0 = load_a_frag(Qb, DHEAD, lane);
  const v16bf aq1 = load_a_frag(Qb + 32, DHEAD, lane);

  const int nl = lane & 15;
  const int kh = lane >> 4;
  const int mh = kh * 8;
  const int qrow0 = qt * 16 + mh;

  float mrow[8], lrow[8];
  v8f o[4] = {};
#pragma unroll
  for (int r = 0; r < 8; ++r) { mrow[r] = -1e30f; lrow[r] = 0.f; }

  for (int kc = 0; kc < NPAD; kc += 32) {
    // ---- hoisted K^T fragments, then 4 score WMMAs ----
    v16bf bk[4];
    bk[0] = load_bT_frag(Kb + (size_t)kc * DHEAD, DHEAD, lane);
    bk[1] = load_bT_frag(Kb + (size_t)kc * DHEAD + 32, DHEAD, lane);
    bk[2] = load_bT_frag(Kb + (size_t)(kc + 16) * DHEAD, DHEAD, lane);
    bk[3] = load_bT_frag(Kb + (size_t)(kc + 16) * DHEAD + 32, DHEAD, lane);
    __builtin_prefetch(Kb + (size_t)(kc + 32 + nl) * DHEAD, 0, 1);
    v8f s0 = {}, s1 = {};
    s0 = wmma_bf16(aq0, bk[0], s0);
    s0 = wmma_bf16(aq1, bk[1], s0);
    s1 = wmma_bf16(aq0, bk[2], s1);
    s1 = wmma_bf16(aq1, bk[3], s1);

    // ---- issue V fragment loads early (hide under softmax) ----
    v16bf bv[4];
#pragma unroll
    for (int j = 0; j < 4; ++j)
      bv[j] = load_bV_frag(Vb + (size_t)kc * DHEAD, j * 16, DHEAD, lane);

    // ---- online softmax over this 32-key chunk ----
    const int key0 = kc + nl, key1 = kc + 16 + nl;
    float p0[8], p1[8], alpha[8];
#pragma unroll
    for (int r = 0; r < 8; ++r) {
      const int q = qrow0 + r;
      float x0 = s0[r] * 0.125f;   // 1/sqrt(64)
      float x1 = s1[r] * 0.125f;
      if (q < SEQ) {
        if (key0 < SEQ) x0 += Bbias[(size_t)q * SEQ + key0];
        if (key1 < SEQ) x1 += Bbias[(size_t)q * SEQ + key1];
      }
      if (key0 >= SEQ) x0 = -1e30f;
      if (key1 >= SEQ) x1 = -1e30f;
      float xm = fmaxf(x0, x1);
#pragma unroll
      for (int off = 8; off > 0; off >>= 1) xm = fmaxf(xm, __shfl_xor(xm, off));
      const float mn = fmaxf(mrow[r], xm);
      const float al = __expf(mrow[r] - mn);
      const float e0 = __expf(x0 - mn);
      const float e1 = __expf(x1 - mn);
      float rs = e0 + e1;
#pragma unroll
      for (int off = 8; off > 0; off >>= 1) rs += __shfl_xor(rs, off);
      lrow[r] = lrow[r] * al + rs;
      mrow[r] = mn;
      alpha[r] = al;
      p0[r] = e0;
      p1[r] = e1;
    }
#pragma unroll
    for (int j = 0; j < 4; ++j)
#pragma unroll
      for (int r = 0; r < 8; ++r) o[j][r] *= alpha[r];

    // ---- transpose P (D-layout) -> A-layout through LDS ----
#pragma unroll
    for (int r = 0; r < 8; ++r) {
      Pst[(mh + r) * 32 + nl] = (bf16_t)p0[r];
      Pst[(mh + r) * 32 + 16 + nl] = (bf16_t)p1[r];
    }
    __syncthreads();
    const bf16_t* pp = &Pst[nl * 32 + kh * 16];
    const v16bf ap = load_frag_2x8(pp, pp + 8);
    __syncthreads();

    // ---- O += P * V: 4 WMMAs ----
#pragma unroll
    for (int j = 0; j < 4; ++j) o[j] = wmma_bf16(ap, bv[j], o[j]);
  }

  // ---- normalize and store to y[b, n, h, d] (token-major, bf16) ----
#pragma unroll
  for (int r = 0; r < 8; ++r) {
    const int q = qrow0 + r;
    if (q < SEQ) {
      const float inv = 1.0f / lrow[r];
      const size_t t = (size_t)(b * SEQ + q) * DMODEL + hh * DHEAD;
#pragma unroll
      for (int j = 0; j < 4; ++j) Y[t + j * 16 + nl] = (bf16_t)(o[j][r] * inv);
    }
  }
}

// ---------------------------------------------------------------------------
// Kernel 5: out = x + sigmoid(gate_lin) * y_proj
// ---------------------------------------------------------------------------
__global__ void k_epilogue(const float* __restrict__ x, const float* __restrict__ gate,
                           const float* __restrict__ yproj, float* __restrict__ out,
                           size_t n) {
  const size_t i = (size_t)blockIdx.x * blockDim.x + threadIdx.x;
  if (i < n) {
    const float sg = 1.0f / (1.0f + __expf(-gate[i]));
    out[i] = x[i] + sg * yproj[i];
  }
}

// ---------------------------------------------------------------------------
// Host-side orchestration
// ---------------------------------------------------------------------------
extern "C" void kernel_launch(void* const* d_in, const int* in_sizes, int n_in,
                              void* d_out, int out_size, void* d_ws, size_t ws_size,
                              hipStream_t stream) {
  (void)in_sizes; (void)n_in; (void)out_size; (void)ws_size;
  const float* x      = (const float*)d_in[0];
  const float* fc     = (const float*)d_in[1];
  const float* g      = (const float*)d_in[2];
  const float* qkv_w  = (const float*)d_in[3];
  const float* qkv_b  = (const float*)d_in[4];
  const float* out_w  = (const float*)d_in[5];
  const float* out_b  = (const float*)d_in[6];
  const float* gate_w = (const float*)d_in[7];
  const float* gate_b = (const float*)d_in[8];
  const float* Bbias  = (const float*)d_in[9];
  float* out = (float*)d_out;

  // workspace carve-out (256B aligned)
  size_t off = 0;
  auto carve = [&](size_t bytes) -> char* {
    char* p = (char*)d_ws + off;
    off += (bytes + 255) & ~(size_t)255;
    return p;
  };
  bf16_t* Wqkv  = (bf16_t*)carve((size_t)3 * DMODEL * DMODEL * 2);
  bf16_t* Wout  = (bf16_t*)carve((size_t)DMODEL * DMODEL * 2);
  bf16_t* Wgate = (bf16_t*)carve((size_t)DMODEL * DMODEL * 2);
  bf16_t* Hbf   = (bf16_t*)carve((size_t)MPAD * DMODEL * 2);
  float*  QKVf  = (float*)carve((size_t)MPAD * 3 * DMODEL * 4);
  bf16_t* Qb    = (bf16_t*)carve((size_t)BATCH * HEADS * NPAD * DHEAD * 2);
  bf16_t* Kb    = (bf16_t*)carve((size_t)BATCH * HEADS * NPAD * DHEAD * 2);
  bf16_t* Vb    = (bf16_t*)carve((size_t)BATCH * HEADS * NPAD * DHEAD * 2);
  bf16_t* Ybf   = (bf16_t*)carve((size_t)MPAD * DMODEL * 2);
  float*  Gf    = (float*)carve((size_t)MPAD * DMODEL * 4);
  float*  Pf    = (float*)carve((size_t)MPAD * DMODEL * 4);

  // 0) weight conversion
  {
    const int n1 = 3 * DMODEL * DMODEL, n2 = DMODEL * DMODEL;
    k_cvt_bf16<<<(n1 + 255) / 256, 256, 0, stream>>>(qkv_w, Wqkv, n1);
    k_cvt_bf16<<<(n2 + 255) / 256, 256, 0, stream>>>(out_w, Wout, n2);
    k_cvt_bf16<<<(n2 + 255) / 256, 256, 0, stream>>>(gate_w, Wgate, n2);
  }
  // 1) zc-RMSNorm (+ zero pad rows of h and y)
  k_zc_rmsnorm<<<MPAD, 256, 0, stream>>>(x, g, Hbf, Ybf);
  // 2) QKV GEMM: [MPAD,1024] x [3072,1024]^T  (TDM-staged weights)
  k_gemm_bf16<<<dim3(3 * DMODEL / BN, MBLK), 128, 0, stream>>>(Hbf, Wqkv, qkv_b, QKVf,
                                                               DMODEL, 3 * DMODEL);
  // 3) RoPE + head split
  {
    const size_t total = (size_t)BATCH * NPAD * (3 * DMODEL);
    k_rope_split<<<(unsigned)((total + 255) / 256), 256, 0, stream>>>(QKVf, fc, Qb, Kb,
                                                                      Vb);
  }
  // 4) flash attention
  k_attention<<<dim3(NQT, HEADS, BATCH), 32, 0, stream>>>(Qb, Kb, Vb, Bbias, Ybf);
  // 5) gate GEMM and out-proj GEMM
  k_gemm_bf16<<<dim3(DMODEL / BN, MBLK), 128, 0, stream>>>(Hbf, Wgate, gate_b, Gf,
                                                           DMODEL, DMODEL);
  k_gemm_bf16<<<dim3(DMODEL / BN, MBLK), 128, 0, stream>>>(Ybf, Wout, out_b, Pf,
                                                           DMODEL, DMODEL);
  // 6) gated residual epilogue
  {
    const size_t n = (size_t)BATCH * SEQ * DMODEL;
    k_epilogue<<<(unsigned)((n + 255) / 256), 256, 0, stream>>>(x, Gf, Pf, out, n);
  }
}
